// SelfAttention_14800457302355
// MI455X (gfx1250) — compile-verified, matrix operand order checked by reading, not run on hardware
//
#include <hip/hip_runtime.h>

// Self-attention, N=8192 tokens, M=1024 dim, fused flash-attention middle.
// Pipeline (all bf16 tensor-core GEMMs, f32 accumulate):
//   1) Kb = bf16(x @ Wk^T)                 [8192,1024]
//   2) Qb = bf16(0.06 * x @ Wq^T)          [8192,1024]
//   3) Vt = bf16(x @ Wv^T)^T               [1024,8192]  (transposed for PV B-frags)
//   4) Yb = flash_attn(Qb,Kb,Vt,u)         [8192,1024] bf16
//   5) out = Yb @ Wo^T                     [8192,1024] f32
// Workspace: 4 x 16MB = 64MB in d_ws.

typedef unsigned short u16t;
typedef __attribute__((ext_vector_type(16))) __bf16 v16bf;
typedef __attribute__((ext_vector_type(8)))  __bf16 bf16x8;
typedef __attribute__((ext_vector_type(8)))  float  v8f;

__device__ __forceinline__ u16t f2bf(float f) {
  unsigned int x = __float_as_uint(f);
  x += 0x7FFFu + ((x >> 16) & 1u);   // round-to-nearest-even
  return (u16t)(x >> 16);
}

__device__ __forceinline__ v8f zero8() {
  v8f z;
#pragma unroll
  for (int i = 0; i < 8; ++i) z[i] = 0.0f;
  return z;
}

// 16-bit A/B fragment (16 rows x 32 k), row-major source with given stride.
// Per ISA 7.12.2: lane = row (M for A, N for B), lane-half selects K sub-block:
//   elements 0..7  -> k = (lane>>4)*8 + 0..7
//   elements 8..15 -> k = 16 + (lane>>4)*8 + 0..7
// Both halves are 16B-contiguous -> two b128 loads.
__device__ __forceinline__ v16bf ld_frag(const u16t* base, int stride, int lane) {
  const int r = lane & 15, h = lane >> 4;
  union { v16bf v; bf16x8 g[2]; } u;
  u.g[0] = *(const bf16x8*)(base + (size_t)r * stride + h * 8);
  u.g[1] = *(const bf16x8*)(base + (size_t)r * stride + 16 + h * 8);
  return u.v;
}

// Same fragment, but the caller already resolved the per-lane row pointer.
__device__ __forceinline__ v16bf ld_frag_row(const u16t* rowptr, int lane) {
  const int h = lane >> 4;
  union { v16bf v; bf16x8 g[2]; } u;
  u.g[0] = *(const bf16x8*)(rowptr + h * 8);
  u.g[1] = *(const bf16x8*)(rowptr + 16 + h * 8);
  return u.v;
}

// C = A[8192,1024] @ W[1024,1024]^T.  128x128 tile per block, 8 waves
// (4x2), each wave does a 32x64 patch = 2x4 WMMA accumulators.
// A_BF16: 0 -> A is f32 (cast while staging), 1 -> A already bf16.
// OUT_MODE: 0 -> bf16 row-major, 1 -> bf16 transposed [1024,8192], 2 -> f32.
template <int A_BF16, int OUT_MODE>
__global__ __launch_bounds__(256, 2) void gemm_xwt(
    const float* __restrict__ Af, const u16t* __restrict__ Ab,
    const float* __restrict__ W, u16t* __restrict__ Ob,
    float* __restrict__ Of, float scale) {
  constexpr int KD = 1024;
  constexpr int AS = 40;  // row stride (ushorts): 80B, 16B-aligned, bank-skewed
  __shared__ u16t As[128 * AS];
  __shared__ u16t Bs[128 * AS];

  const int tid = threadIdx.x;
  const int lane = tid & 31, wid = tid >> 5;
  const int m0 = blockIdx.x * 128, n0 = blockIdx.y * 128;
  const int lr = tid >> 1, lc = (tid & 1) * 16;  // staging: 16 elems/thread
  const int r0 = (wid & 3) * 32, c0 = (wid >> 2) * 64;

  v8f acc[2][4];
#pragma unroll
  for (int a = 0; a < 2; ++a)
#pragma unroll
    for (int b = 0; b < 4; ++b) acc[a][b] = zero8();

  for (int k0 = 0; k0 < KD; k0 += 32) {
    if (A_BF16) {
      const u16t* s = Ab + (size_t)(m0 + lr) * KD + k0 + lc;
#pragma unroll
      for (int e = 0; e < 16; ++e) As[lr * AS + lc + e] = s[e];
    } else {
      const float* s = Af + (size_t)(m0 + lr) * KD + k0 + lc;
#pragma unroll
      for (int e = 0; e < 16; ++e) As[lr * AS + lc + e] = f2bf(s[e]);
    }
    {
      // Bs[n][k] = W[n0+n][k0+k]  (W row-major == k-contiguous: ideal)
      const float* s = W + (size_t)(n0 + lr) * KD + k0 + lc;
#pragma unroll
      for (int e = 0; e < 16; ++e) Bs[lr * AS + lc + e] = f2bf(s[e]);
    }
    __syncthreads();

    v16bf a0 = ld_frag(&As[(r0 + 0) * AS], AS, lane);
    v16bf a1 = ld_frag(&As[(r0 + 16) * AS], AS, lane);
#pragma unroll
    for (int fn = 0; fn < 4; ++fn) {
      v16bf b = ld_frag(&Bs[(c0 + fn * 16) * AS], AS, lane);
      acc[0][fn] = __builtin_amdgcn_wmma_f32_16x16x32_bf16(
          false, a0, false, b, (short)0, acc[0][fn], false, false);
      acc[1][fn] = __builtin_amdgcn_wmma_f32_16x16x32_bf16(
          false, a1, false, b, (short)0, acc[1][fn], false, false);
    }
    __syncthreads();
  }

  // D layout: element i of v8f -> m = (lane>>4)*8 + i, n = lane&15
  const int n = lane & 15, h = lane >> 4;
#pragma unroll
  for (int fr = 0; fr < 2; ++fr)
#pragma unroll
    for (int fn = 0; fn < 4; ++fn)
#pragma unroll
      for (int i = 0; i < 8; ++i) {
        const int row = m0 + r0 + fr * 16 + h * 8 + i;
        const int col = n0 + c0 + fn * 16 + n;
        const float v = acc[fr][fn][i] * scale;
        if (OUT_MODE == 0)
          Ob[(size_t)row * 1024 + col] = f2bf(v);
        else if (OUT_MODE == 1)
          Ob[(size_t)col * 8192 + row] = f2bf(v);
        else
          Of[(size_t)row * 1024 + col] = v;
      }
}

// Flash attention: block = 16 query rows, 8 waves. Q tile resident in LDS.
// Loop over keys in tiles of 128: wave w computes S columns [w*16,w*16+16)
// (K B-frags straight from global: row-major K == k-contiguous), then online
// softmax + dropout (keep folded into P), then each wave accumulates its own
// 128-wide output column slice via P@V with Vt B-frags from global.
__global__ __launch_bounds__(256, 1) void attn_kernel(
    const u16t* __restrict__ Qb, const u16t* __restrict__ Kb,
    const u16t* __restrict__ Vt, const float* __restrict__ U,
    u16t* __restrict__ Yb) {
  constexpr int QS = 1032;  // 2064B row stride, 16B aligned, skewed
  constexpr int SS = 129;
  constexpr int PS = 144;   // 288B row stride, 16B aligned
  __shared__ u16t Qs[16 * QS];
  __shared__ float Ssm[16 * SS];
  __shared__ u16t Psm[16 * PS];
  __shared__ float m_s[16], l_s[16], sc_s[16];

  const int tid = threadIdx.x, lane = tid & 31, wid = tid >> 5;
  const int q0 = blockIdx.x * 16;
  const int nn = lane & 15, h = lane >> 4;

  for (int idx = tid; idx < 16 * 1024; idx += 256) {
    const int r = idx >> 10, c = idx & 1023;
    Qs[r * QS + c] = Qb[(size_t)(q0 + r) * 1024 + c];
  }
  if (tid < 16) { m_s[tid] = -3.0e38f; l_s[tid] = 0.0f; }
  v8f acc[8];
#pragma unroll
  for (int f = 0; f < 8; ++f) acc[f] = zero8();
  __syncthreads();

  for (int j0 = 0; j0 < 8192; j0 += 128) {
    // Phase 1: S[0:16, wid*16 : wid*16+16] over full 1024 k-dim
    v8f s = zero8();
    const u16t* krow = Kb + (size_t)(j0 + wid * 16 + nn) * 1024;
#pragma unroll 4
    for (int kc = 0; kc < 1024; kc += 32) {
      v16bf a = ld_frag(Qs + kc, QS, lane);
      v16bf b = ld_frag_row(krow + kc, lane);
      s = __builtin_amdgcn_wmma_f32_16x16x32_bf16(false, a, false, b,
                                                  (short)0, s, false, false);
    }
#pragma unroll
    for (int i = 0; i < 8; ++i) Ssm[(h * 8 + i) * SS + wid * 16 + nn] = s[i];
    __syncthreads();

    // Phase 2a: per-row running max + rescale factor
    if (tid < 16) {
      float mold = m_s[tid], mx = mold;
      const float* row = &Ssm[tid * SS];
      for (int j = 0; j < 128; ++j) mx = fmaxf(mx, row[j]);
      sc_s[tid] = __expf(mold - mx);
      m_s[tid] = mx;
    }
    __syncthreads();

    // Phase 2b: e = exp(S - m); P = (u >= 0.5 ? e : 0) in bf16.
    // u is streamed exactly once -> non-temporal so it can't evict the
    // L2-resident K/Q/Vt working set (48MB << 192MB L2).
    for (int idx = tid; idx < 2048; idx += 256) {
      const int i = idx >> 7, j = idx & 127;
      const float e = __expf(Ssm[i * SS + j] - m_s[i]);
      Ssm[i * SS + j] = e;
      const float uu =
          __builtin_nontemporal_load(U + (size_t)(q0 + i) * 8192 + j0 + j);
      Psm[i * PS + j] = f2bf(uu >= 0.5f ? e : 0.0f);
    }
    __syncthreads();

    // Phase 2c: l = l*sc + sum(e)   (denominator uses pre-dropout probs)
    if (tid < 16) {
      float l = l_s[tid] * sc_s[tid];
      const float* row = &Ssm[tid * SS];
      for (int j = 0; j < 128; ++j) l += row[j];
      l_s[tid] = l;
    }
    __syncthreads();

    // Phase 3: acc *= sc ; acc += P @ V  (wave owns cols [wid*128, +128))
    float scs[8];
#pragma unroll
    for (int i = 0; i < 8; ++i) scs[i] = sc_s[h * 8 + i];
#pragma unroll
    for (int f = 0; f < 8; ++f)
#pragma unroll
      for (int i = 0; i < 8; ++i) acc[f][i] *= scs[i];

#pragma unroll
    for (int jc = 0; jc < 128; jc += 32) {
      v16bf a = ld_frag(Psm + jc, PS, lane);
#pragma unroll
      for (int f = 0; f < 8; ++f) {
        const u16t* vrow =
            Vt + (size_t)(wid * 128 + f * 16 + nn) * 8192 + j0 + jc;
        v16bf b = ld_frag_row(vrow, lane);
        acc[f] = __builtin_amdgcn_wmma_f32_16x16x32_bf16(
            false, a, false, b, (short)0, acc[f], false, false);
      }
    }
    __syncthreads();
  }

  // y = acc * (2/l)   (2x = dropout 1/(1-p) folded here)
  float inv[8];
#pragma unroll
  for (int i = 0; i < 8; ++i) inv[i] = 2.0f / l_s[h * 8 + i];
#pragma unroll
  for (int f = 0; f < 8; ++f)
#pragma unroll
    for (int i = 0; i < 8; ++i) {
      const int row = q0 + h * 8 + i;
      const int col = wid * 128 + f * 16 + nn;
      Yb[(size_t)row * 1024 + col] = f2bf(acc[f][i] * inv[i]);
    }
}

extern "C" void kernel_launch(void* const* d_in, const int* in_sizes, int n_in,
                              void* d_out, int out_size, void* d_ws,
                              size_t ws_size, hipStream_t stream) {
  const float* x  = (const float*)d_in[0];
  const float* Wk = (const float*)d_in[1];
  const float* Wq = (const float*)d_in[2];
  const float* Wv = (const float*)d_in[3];
  const float* Wo = (const float*)d_in[4];
  const float* U  = (const float*)d_in[5];
  float* out = (float*)d_out;

  // workspace layout: Kb | Qb | Vt | Yb, each 8192*1024 bf16 = 16MB (64MB total)
  u16t* Kb = (u16t*)d_ws;
  u16t* Qb = Kb + (size_t)8192 * 1024;
  u16t* Vt = Qb + (size_t)8192 * 1024;
  u16t* Yb = Vt + (size_t)8192 * 1024;

  dim3 g(64, 8), b(256);
  gemm_xwt<0, 0><<<g, b, 0, stream>>>(x, nullptr, Wk, Kb, nullptr, 1.0f);
  gemm_xwt<0, 0><<<g, b, 0, stream>>>(x, nullptr, Wq, Qb, nullptr, 0.06f);
  gemm_xwt<0, 1><<<g, b, 0, stream>>>(x, nullptr, Wv, Vt, nullptr, 1.0f);
  attn_kernel<<<dim3(512), b, 0, stream>>>(Qb, Kb, Vt, U, Yb);
  gemm_xwt<1, 2><<<g, b, 0, stream>>>(nullptr, Yb, Wo, nullptr, out, 1.0f);
}
